// MultiHeadSelfAttention_25993142075424
// MI455X (gfx1250) — matrix operand order repaired and checked by reading, not compiled, not run on hardware
//
#include <hip/hip_runtime.h>

typedef __attribute__((ext_vector_type(16))) _Float16 v16h;
typedef __attribute__((ext_vector_type(8)))  _Float16 v8h;
typedef __attribute__((ext_vector_type(2)))  _Float16 v2h;
typedef __attribute__((ext_vector_type(8)))  float    v8f;
typedef __attribute__((ext_vector_type(4)))  float    v4f;
typedef __attribute__((ext_vector_type(4)))  unsigned int v4u;
typedef __attribute__((ext_vector_type(8)))  int      v8i;
typedef __attribute__((ext_vector_type(4)))  int      v4i;

#define D_MODEL   768
#define NUM_HEADS 12
#define HEAD_DIM  64
#define BATCH     2
#define SEQ       2048
#define MTOT      (BATCH * SEQ)   // 4096

__device__ __forceinline__ v8f wmma_f16(v16h a, v16h b, v8f c) {
  // D = A(16x32 f16) x B(32x16 f16) + C(16x16 f32)
  return __builtin_amdgcn_wmma_f32_16x16x32_f16(false, a, false, b, (short)0, c,
                                                false, false);
}

// ---- TDM: async 2-D tile load Global -> LDS (D# per ISA 08_async_tensor §8) -
// tile_d0 elements contiguous (data_size=2B), tile_d1 lines of stride0 elems.
__device__ __forceinline__ void tdm_load_2d(unsigned lds_off, const void* gptr,
                                            unsigned tile_d0, unsigned tile_d1,
                                            unsigned tdim0, unsigned tdim1,
                                            unsigned stride0) {
  unsigned long long ga = (unsigned long long)gptr;
  v4u g0;
  g0[0] = 1u;                                         // count=1, user mode
  g0[1] = lds_off;                                    // lds_addr (bytes)
  g0[2] = (unsigned)ga;                               // global_addr[31:0]
  g0[3] = ((unsigned)(ga >> 32) & 0x01FFFFFFu) | 0x80000000u;  // addr[56:32]|type=2
  v8i g1;
  g1[0] = 0x10000;                                    // data_size=1 -> 2 bytes
  g1[1] = (int)(tdim0 << 16);                         // tensor_dim0[15:0]
  g1[2] = (int)((tdim0 >> 16) | (tdim1 << 16));       // tdim0 hi | tdim1 lo
  g1[3] = (int)((tdim1 >> 16) | (tile_d0 << 16));     // tdim1 hi | tile_dim0
  g1[4] = (int)(tile_d1 & 0xFFFFu);                   // tile_dim1 (tile_dim2=0)
  g1[5] = (int)stride0;                               // tensor_dim0_stride lo
  g1[6] = 0;                                          // stride0 hi | dim1_stride lo
  g1[7] = 0;
  v4i z4 = {0, 0, 0, 0};
  v8i z8 = {0, 0, 0, 0, 0, 0, 0, 0};
  __builtin_amdgcn_tensor_load_to_lds(g0, g1, z4, z4, z8, 0);
}

__device__ __forceinline__ unsigned lds_offset(const void* p) {
  return (unsigned)(unsigned long long)p;             // generic LDS ptr: low 32 bits
}

// ---- fragment loaders -------------------------------------------------------
// A-frag 16x32 from fp32 row-major. Lane L: M=L&15, K = kb + {0..7, 16..23},
// kb = (L&16)?8:0  (ISA 7.12.2 16-bit A layout).
__device__ __forceinline__ v16h load_a_f32(const float* __restrict__ base, int lda,
                                           int m0, int k0, int lane) {
  int m  = lane & 15;
  int kb = (lane & 16) ? 8 : 0;
  const float* p = base + (size_t)(m0 + m) * lda + k0 + kb;
  v4f x0 = *(const v4f*)(p);
  v4f x1 = *(const v4f*)(p + 4);
  v4f x2 = *(const v4f*)(p + 16);
  v4f x3 = *(const v4f*)(p + 20);
  v16h a;
#pragma unroll
  for (int i = 0; i < 4; ++i) {
    a[i]      = (_Float16)x0[i];
    a[4 + i]  = (_Float16)x1[i];
    a[8 + i]  = (_Float16)x2[i];
    a[12 + i] = (_Float16)x3[i];
  }
  return a;
}

// A-frag 16x32 from f16 row-major (lda halves).
__device__ __forceinline__ v16h load_a_f16(const _Float16* __restrict__ base, int lda,
                                           int m0, int k0, int lane) {
  int m  = lane & 15;
  int kb = (lane & 16) ? 8 : 0;
  const _Float16* p = base + (size_t)(m0 + m) * lda + k0 + kb;
  v8h x0 = *(const v8h*)(p);
  v8h x1 = *(const v8h*)(p + 16);
  v16h a;
#pragma unroll
  for (int i = 0; i < 8; ++i) { a[i] = x0[i]; a[8 + i] = x1[i]; }
  return a;
}

// B-frag 32x16 where B[k][n] = Src[n0+n][k0+k], Src fp32 row-major.
// Lane L: N=L&15, K = kb + {0..7, 16..23}, kb = (L&16)?8:0 (matches A layout).
__device__ __forceinline__ v16h load_b_wf32(const float* __restrict__ W, int ldk,
                                            int n0, int k0, int lane) {
  int n  = lane & 15;
  int kb = (lane & 16) ? 8 : 0;
  const float* p = W + (size_t)(n0 + n) * ldk + k0 + kb;
  v4f x0 = *(const v4f*)(p);
  v4f x1 = *(const v4f*)(p + 4);
  v4f x2 = *(const v4f*)(p + 16);
  v4f x3 = *(const v4f*)(p + 20);
  v16h b;
#pragma unroll
  for (int i = 0; i < 4; ++i) {
    b[i]      = (_Float16)x0[i];
    b[4 + i]  = (_Float16)x1[i];
    b[8 + i]  = (_Float16)x2[i];
    b[12 + i] = (_Float16)x3[i];
  }
  return b;
}

// ---- kernel 1: QKV projection  out = X @ W^T + b ----------------------------
// Wave computes a 32x64 output tile: 2 A-frags x 4 B-frags -> 8 WMMAs / K-step.
// VT=false: write f16 [B,H,S,Dh] (Q,K). VT=true: write f16 [B,H,Dh,S] (V^T).
template <bool VT>
__global__ __launch_bounds__(128) void qkv_proj_kernel(
    const float* __restrict__ X, const float* __restrict__ W,
    const float* __restrict__ bias, _Float16* __restrict__ out) {
  int wave = (blockIdx.x * blockDim.x + threadIdx.x) >> 5;
  int lane = threadIdx.x & 31;
  int tn   = wave % (D_MODEL / 64);
  int tm   = wave / (D_MODEL / 64);
  int m0 = tm * 32, n0 = tn * 64;

  v8f c[2][4] = {};
  for (int k0 = 0; k0 < D_MODEL; k0 += 32) {
    if (k0 + 32 < D_MODEL) {
      __builtin_prefetch(X + (size_t)m0 * D_MODEL + k0 + 32, 0, 1);
      __builtin_prefetch(W + (size_t)n0 * D_MODEL + k0 + 32, 0, 1);
    }
    v16h a0 = load_a_f32(X, D_MODEL, m0, k0, lane);
    v16h a1 = load_a_f32(X, D_MODEL, m0 + 16, k0, lane);
#pragma unroll
    for (int t = 0; t < 4; ++t) {
      v16h b = load_b_wf32(W, D_MODEL, n0 + 16 * t, k0, lane);
      c[0][t] = wmma_f16(a0, b, c[0][t]);
      c[1][t] = wmma_f16(a1, b, c[1][t]);
    }
  }

  int ncol = lane & 15;
  int off  = (lane & 16) ? 8 : 0;
#pragma unroll
  for (int t = 0; t < 4; ++t) {
    int n   = n0 + 16 * t + ncol;
    float bv = bias[n];
    int h = n >> 6, d = n & 63;
#pragma unroll
    for (int i = 0; i < 2; ++i) {
      if (VT) {
        int mb = m0 + 16 * i + off;             // 8 consecutive global rows
        int bi = mb >> 11, s0 = mb & (SEQ - 1);
        v8h hv;
#pragma unroll
        for (int r = 0; r < 8; ++r) hv[r] = (_Float16)(c[i][t][r] + bv);
        *(v8h*)(&out[(((size_t)(bi * NUM_HEADS + h) * HEAD_DIM + d) * SEQ) + s0]) = hv;
      } else {
#pragma unroll
        for (int r = 0; r < 8; ++r) {
          int m = m0 + 16 * i + off + r;
          int bi = m >> 11, s = m & (SEQ - 1);
          out[(((size_t)(bi * NUM_HEADS + h) * SEQ + s) * HEAD_DIM) + d] =
              (_Float16)(c[i][t][r] + bv);
        }
      }
    }
  }
}

// ---- kernel 2: causal flash attention ---------------------------------------
// Block = 4 waves = 64 consecutive query rows of one (b,h). K/V chunks (32 kv)
// are DMA'd into LDS by the Tensor Data Mover (wave 0), double-buffered, and
// shared by all 4 waves. Per-wave: 16-row query tile, online softmax.
__global__ __launch_bounds__(128) void attn_kernel(
    const _Float16* __restrict__ q, const _Float16* __restrict__ k,
    const _Float16* __restrict__ vt, _Float16* __restrict__ ctx) {
  __shared__ _Float16 Ks[2][32][HEAD_DIM];          // [buf][kv][d]
  __shared__ _Float16 Vs[2][HEAD_DIM][32];          // [buf][d][kv]
  __shared__ alignas(16) v2h pshare[4][16][16];     // P packed: [m][c]={P[m][c],P[m][c+16]}

  int lane = threadIdx.x & 31;
  int wib  = threadIdx.x >> 5;
  int qb   = blockIdx.x & (SEQ / 64 - 1);     // 32 query blocks per (b,h)
  int bh   = blockIdx.x >> 5;                 // [0, B*H)
  int q0b  = qb * 64;
  int q0   = q0b + wib * 16;

  const _Float16* Q  = q  + (size_t)bh * SEQ * HEAD_DIM;
  const _Float16* K  = k  + (size_t)bh * SEQ * HEAD_DIM;
  const _Float16* Vt = vt + (size_t)bh * HEAD_DIM * SEQ;   // [Dh][S]

  v16h qa0 = load_a_f16(Q, HEAD_DIM, q0, 0, lane);
  v16h qa1 = load_a_f16(Q, HEAD_DIM, q0, 32, lane);

  v16h onesB;                       // all-ones B-frag: row-sum via WMMA
#pragma unroll
  for (int i = 0; i < 16; ++i) onesB[i] = (_Float16)1.0f;

  int off  = (lane & 16) ? 8 : 0;   // C-frag row offset
  int ncol = lane & 15;             // C-frag column
  float mrow[8], lrow[8];
  v8f acc[4] = {};
#pragma unroll
  for (int r = 0; r < 8; ++r) { mrow[r] = -__builtin_inff(); lrow[r] = 0.f; }

  const float scale = 0.125f;       // 1/sqrt(64)
  v2h* pl2 = &pshare[wib][0][0];

  int trips = qb * 2 + 2;           // kv chunks of 32 covering kv < q0b+64

  if (wib == 0) {                   // stage chunk 0
    tdm_load_2d(lds_offset(&Ks[0][0][0]), K, HEAD_DIM, 32, HEAD_DIM, SEQ, HEAD_DIM);
    tdm_load_2d(lds_offset(&Vs[0][0][0]), Vt, 32, HEAD_DIM, SEQ, HEAD_DIM, SEQ);
  }

  for (int it = 0; it < trips; ++it) {
    int kv0 = it * 32;
    int cur = it & 1;
    if (wib == 0) {
      if (it + 1 < trips) {         // prefetch next chunk into other buffer
        int kvn = kv0 + 32;
        tdm_load_2d(lds_offset(&Ks[cur ^ 1][0][0]), K + (size_t)kvn * HEAD_DIM,
                    HEAD_DIM, 32, HEAD_DIM, SEQ, HEAD_DIM);
        tdm_load_2d(lds_offset(&Vs[cur ^ 1][0][0]), Vt + kvn,
                    32, HEAD_DIM, SEQ, HEAD_DIM, SEQ);
        __builtin_amdgcn_s_wait_tensorcnt(2);   // chunk `it` complete
      } else {
        __builtin_amdgcn_s_wait_tensorcnt(0);
      }
    }
    __syncthreads();                // publish buffer `cur` to all waves

    // ---- scores: S = Q K^T from LDS (B[k=d][n=kv] = Ks[kv][d]) ----
    v16h kb0lo, kb0hi, kb1lo, kb1hi;
    {
      int kbB = (lane & 16) ? 8 : 0;
      const v8h* p0 = (const v8h*)&Ks[cur][ncol][kbB];
      const v8h* p1 = (const v8h*)&Ks[cur][ncol][32 + kbB];
      const v8h* p2 = (const v8h*)&Ks[cur][ncol + 16][kbB];
      const v8h* p3 = (const v8h*)&Ks[cur][ncol + 16][32 + kbB];
      v8h a0 = p0[0], a1 = p0[2], b0 = p1[0], b1 = p1[2];
      v8h c0 = p2[0], c1 = p2[2], d0 = p3[0], d1 = p3[2];
#pragma unroll
      for (int i = 0; i < 8; ++i) {
        kb0lo[i] = a0[i]; kb0lo[8 + i] = a1[i];
        kb0hi[i] = b0[i]; kb0hi[8 + i] = b1[i];
        kb1lo[i] = c0[i]; kb1lo[8 + i] = c1[i];
        kb1hi[i] = d0[i]; kb1hi[8 + i] = d1[i];
      }
    }
    v8f s0 = {};
    s0 = wmma_f16(qa0, kb0lo, s0);
    s0 = wmma_f16(qa1, kb0hi, s0);
    v8f s1 = {};
    s1 = wmma_f16(qa0, kb1lo, s1);
    s1 = wmma_f16(qa1, kb1hi, s1);

    // ---- online softmax (fully-masked chunks are inert: corr=1, p=0) ----
    float corr[8];
#pragma unroll
    for (int r = 0; r < 8; ++r) {
      int qm = q0 + r + off;
      float a0 = (kv0 + ncol      <= qm) ? s0[r] * scale : -__builtin_inff();
      float a1 = (kv0 + 16 + ncol <= qm) ? s1[r] * scale : -__builtin_inff();
      float t = fmaxf(a0, a1);                 // row max across 16 lanes of half
      t = fmaxf(t, __shfl_xor(t, 1, 32));
      t = fmaxf(t, __shfl_xor(t, 2, 32));
      t = fmaxf(t, __shfl_xor(t, 4, 32));
      t = fmaxf(t, __shfl_xor(t, 8, 32));
      float nm = fmaxf(mrow[r], t);
      corr[r]  = __expf(mrow[r] - nm);
      mrow[r]  = nm;
      float p0 = __expf(a0 - nm);
      float p1 = __expf(a1 - nm);
#pragma unroll
      for (int t4 = 0; t4 < 4; ++t4) acc[t4][r] *= corr[r];
      v2h pp; pp[0] = (_Float16)p0; pp[1] = (_Float16)p1;
      pl2[(r + off) * 16 + ncol] = pp;          // packed P word
    }

    // re-read P in A-fragment layout from packed words (2x ds_load_b128)
    v16h pa;
    {
      int m  = lane & 15;
      int kb = (lane & 16) ? 8 : 0;
      const v8h* pw = (const v8h*)&pl2[m * 16 + kb];
      v8h w0 = pw[0], w1 = pw[1];
#pragma unroll
      for (int e = 0; e < 4; ++e) {
        pa[e]      = w0[2 * e];      // P[m][kb+e]
        pa[8 + e]  = w0[2 * e + 1];  // P[m][kb+16+e]
        pa[4 + e]  = w1[2 * e];      // P[m][kb+4+e]
        pa[12 + e] = w1[2 * e + 1];  // P[m][kb+20+e]
      }
    }

    // row sums via WMMA with all-ones B (replaces 32 shuffle-adds)
    v8f rs = {};
    rs = wmma_f16(pa, onesB, rs);
#pragma unroll
    for (int r = 0; r < 8; ++r) lrow[r] = lrow[r] * corr[r] + rs[r];

    // ---- O += P @ V from LDS (B[k=kv][n=d] = Vs[d][kv]) ----
    {
      int kbB = (lane & 16) ? 8 : 0;
#pragma unroll
      for (int t4 = 0; t4 < 4; ++t4) {
        const v8h* pv = (const v8h*)&Vs[cur][t4 * 16 + ncol][kbB];
        v8h x0 = pv[0], x1 = pv[2];
        v16h vb;
#pragma unroll
        for (int i = 0; i < 8; ++i) { vb[i] = x0[i]; vb[8 + i] = x1[i]; }
        acc[t4] = wmma_f16(pa, vb, acc[t4]);
      }
    }
    __syncthreads();                // all waves done with buffer `cur`
  }

  // normalize and write context f16 [B,S,D]
  int bi = bh / NUM_HEADS, h = bh % NUM_HEADS;
#pragma unroll
  for (int r = 0; r < 8; ++r) {
    float inv = 1.f / lrow[r];
    int m = q0 + r + off;
    size_t rowb = ((size_t)(bi * SEQ + m)) * D_MODEL + h * HEAD_DIM;
#pragma unroll
    for (int t4 = 0; t4 < 4; ++t4)
      ctx[rowb + t4 * 16 + ncol] = (_Float16)(acc[t4][r] * inv);
  }
}

// ---- kernel 3: output projection  out = ctx @ Wo^T + bo  (fp32 out) ---------
__global__ __launch_bounds__(128) void out_proj_kernel(
    const _Float16* __restrict__ ctx, const float* __restrict__ W,
    const float* __restrict__ bias, float* __restrict__ out) {
  int wave = (blockIdx.x * blockDim.x + threadIdx.x) >> 5;
  int lane = threadIdx.x & 31;
  int tn   = wave % (D_MODEL / 64);
  int tm   = wave / (D_MODEL / 64);
  int m0 = tm * 32, n0 = tn * 64;

  v8f c[2][4] = {};
  for (int k0 = 0; k0 < D_MODEL; k0 += 32) {
    if (k0 + 32 < D_MODEL) {
      __builtin_prefetch(ctx + (size_t)m0 * D_MODEL + k0 + 32, 0, 1);
      __builtin_prefetch(W + (size_t)n0 * D_MODEL + k0 + 32, 0, 1);
    }
    v16h a0 = load_a_f16(ctx, D_MODEL, m0, k0, lane);
    v16h a1 = load_a_f16(ctx, D_MODEL, m0 + 16, k0, lane);
#pragma unroll
    for (int t = 0; t < 4; ++t) {
      v16h b = load_b_wf32(W, D_MODEL, n0 + 16 * t, k0, lane);
      c[0][t] = wmma_f16(a0, b, c[0][t]);
      c[1][t] = wmma_f16(a1, b, c[1][t]);
    }
  }

  int ncol = lane & 15;
  int off  = (lane & 16) ? 8 : 0;
#pragma unroll
  for (int t = 0; t < 4; ++t) {
    int n = n0 + 16 * t + ncol;
    float bv = bias[n];
#pragma unroll
    for (int i = 0; i < 2; ++i) {
#pragma unroll
      for (int r = 0; r < 8; ++r) {
        int m = m0 + 16 * i + off + r;
        out[(size_t)m * D_MODEL + n] = c[i][t][r] + bv;
      }
    }
  }
}

// ---- host launch ------------------------------------------------------------
extern "C" void kernel_launch(void* const* d_in, const int* in_sizes, int n_in,
                              void* d_out, int out_size, void* d_ws, size_t ws_size,
                              hipStream_t stream) {
  const float* query = (const float*)d_in[0];
  const float* key   = (const float*)d_in[1];
  const float* value = (const float*)d_in[2];
  // d_in[3] is the causal mask; causality is hardcoded in attn_kernel.
  const float* Wq = (const float*)d_in[4];
  const float* bq = (const float*)d_in[5];
  const float* Wk = (const float*)d_in[6];
  const float* bk = (const float*)d_in[7];
  const float* Wv = (const float*)d_in[8];
  const float* bv = (const float*)d_in[9];
  const float* Wo = (const float*)d_in[10];
  const float* bo = (const float*)d_in[11];
  float* out = (float*)d_out;

  _Float16* qws = (_Float16*)d_ws;                       // [B,H,S,Dh] f16
  _Float16* kws = qws + (size_t)MTOT * D_MODEL;          // [B,H,S,Dh] f16
  _Float16* vws = kws + (size_t)MTOT * D_MODEL;          // [B,H,Dh,S] f16 (V^T)
  _Float16* cws = vws + (size_t)MTOT * D_MODEL;          // context [B,S,D] f16

  dim3 blk(128);
  int gemm_waves = (MTOT / 32) * (D_MODEL / 64);         // 1536
  dim3 gemm_grid(gemm_waves / 4);                        // 384 blocks
  qkv_proj_kernel<false><<<gemm_grid, blk, 0, stream>>>(query, Wq, bq, qws);
  qkv_proj_kernel<false><<<gemm_grid, blk, 0, stream>>>(key,   Wk, bk, kws);
  qkv_proj_kernel<true ><<<gemm_grid, blk, 0, stream>>>(value, Wv, bv, vws);

  int attn_blocks = BATCH * NUM_HEADS * (SEQ / 64);      // 768
  attn_kernel<<<dim3(attn_blocks), blk, 0, stream>>>(qws, kws, vws, cws);

  out_proj_kernel<<<gemm_grid, blk, 0, stream>>>(cws, Wo, bo, out);
}
